// SelfAttention_42880953483701
// MI455X (gfx1250) — matrix operand order repaired and checked by reading, not compiled, not run on hardware
//
#include <hip/hip_runtime.h>
#include <cstdint>

// ---------------------------------------------------------------------------
// Self-attention for MI455X (gfx1250): bf16 WMMA + Tensor Data Mover staging.
//   S=2048, B=4, H=1024, NH=16, HD=64, SCALE=8 (folded into Q as 0.125)
// Pipeline:
//   cvt      : f32 -> bf16 copies of query/key/value and Wq/Wk/Wv/Wo  (ws)
//   gemm<Q>  : q = (xq @ Wq^T + bq) * 0.125  -> ws (B,NH,S,HD)  bf16
//   gemm<K>  : k =  xk @ Wk^T + bk           -> ws (B,NH,S,HD)  bf16
//   gemm<V>  : v =  xv @ Wv^T + bv           -> ws (B,NH,HD,S)  bf16 (transposed)
//   flash    : online-softmax attention      -> ws (S,B,H)      bf16
//   gemm<O>  : out = o @ Wo^T + bo           -> d_out (S,B,H)   f32
// GEMMs stage 128x32 bf16 tiles into LDS with TENSOR_LOAD_TO_LDS (TDM),
// double-buffered on TENSORcnt; the D# pad feature reproduces the padded
// LDS row stride (32+8 bf16) the WMMA fragment loads expect.
// ---------------------------------------------------------------------------

#define S_LEN 2048
#define B_SZ  4
#define H_DIM 1024
#define NH_   16
#define HD_   64
#define M_TOT (S_LEN * B_SZ) /* 8192 */

typedef __attribute__((ext_vector_type(16))) __bf16 v16bf;
typedef __attribute__((ext_vector_type(8)))  __bf16 v8bf;
typedef __attribute__((ext_vector_type(4)))  __bf16 v4bf;
typedef __attribute__((ext_vector_type(8)))  float  v8f;
typedef __attribute__((ext_vector_type(4)))  unsigned int u32x4;
typedef __attribute__((ext_vector_type(8)))  int      i32x8;
typedef __attribute__((ext_vector_type(4)))  int      i32x4;

static __device__ __forceinline__ v16bf cat16(v8bf lo, v8bf hi) {
  return __builtin_shufflevector(lo, hi, 0,1,2,3,4,5,6,7,8,9,10,11,12,13,14,15);
}
static __device__ __forceinline__ v16bf ldg16(const __bf16* p) {
  return cat16(*(const v8bf*)p, *(const v8bf*)(p + 8));
}
static __device__ __forceinline__ v8f zero8() {
  v8f z; for (int i = 0; i < 8; i++) z[i] = 0.0f; return z;
}

// ---------------------------------------------------------------------------
// f32 -> bf16 streaming conversion (float4 / v4bf vectorized)
// ---------------------------------------------------------------------------
__global__ __launch_bounds__(256) void cvt_bf16(const float* __restrict__ in,
                                                __bf16* __restrict__ out, int n4) {
  int i = blockIdx.x * 256 + threadIdx.x;
  if (i < n4) {
    float4 f = ((const float4*)in)[i];
    v4bf v;
    v[0] = (__bf16)f.x; v[1] = (__bf16)f.y; v[2] = (__bf16)f.z; v[3] = (__bf16)f.w;
    ((v4bf*)out)[i] = v;
  }
}

// ---------------------------------------------------------------------------
// TDM: DMA one [tile_d1 rows x 32 bf16] tile (row stride = K elements) from
// global into LDS at lds_off, inserting 16B of pad after every 64B row
// (pad_interval=3 -> 16 DWORDs, pad_amount=3 -> 4 DWORDs)  => LDS row stride
// 80B = 40 bf16, matching the fragment loads below.
// D# packing per CDNA5 ISA.
// This toolchain exposes the 6-arg builtin:
//   (uint32x4 g0, int32x8 g1, int32x4 g2, int32x4 g3, int32x8 extra, i32 cpol)
// ---------------------------------------------------------------------------
#define LDSS 40

#if __has_builtin(__builtin_amdgcn_tensor_load_to_lds)
#define USE_TDM 1
#else
#define USE_TDM 0
#endif

static __device__ __forceinline__ void tdm_load_tile(unsigned lds_off,
                                                     const __bf16* gp,
                                                     unsigned rows,
                                                     unsigned stride_elems) {
#if USE_TDM
  unsigned long long ga = (unsigned long long)(uintptr_t)gp;
  u32x4 g0;
  g0[0] = 1u;                                      // count=1, user descriptor
  g0[1] = lds_off;                                 // lds_addr (bytes)
  g0[2] = (unsigned)(ga & 0xFFFFFFFFu);            // global_addr[31:0]
  g0[3] = (unsigned)((ga >> 32) & 0x01FFFFFFu)     // global_addr[56:32]
          | (2u << 30);                            // type=2 ("image")
  i32x8 g1;
  g1[0] = (int)((1u << 16)                         // data_size = 2 bytes
                | (1u << 20)                       // pad_enable
                | (3u << 22)                       // pad_interval: 16 DWORDs
                | (3u << 25));                     // pad_amount:   4 DWORDs
  g1[1] = (int)((stride_elems & 0xFFFFu) << 16);   // tensor_dim0[15:0]
  g1[2] = (int)(((stride_elems >> 16) & 0xFFFFu)   // tensor_dim0[31:16]
                | ((0x100000u & 0xFFFFu) << 16));  // tensor_dim1[15:0] of 1<<20
  g1[3] = (int)(((0x100000u >> 16) & 0xFFFFu)      // tensor_dim1[31:16]
                | (32u << 16));                    // tile_dim0 = 32 elems
  g1[4] = (int)(rows & 0xFFFFu);                   // tile_dim1; tile_dim2=0
  g1[5] = (int)stride_elems;                       // tensor_dim0_stride[31:0]
  g1[6] = 0;                                       // stride hi / dim1_stride lo
  g1[7] = 0;
  i32x4 z4; z4[0] = 0; z4[1] = 0; z4[2] = 0; z4[3] = 0;
  i32x8 z8; for (int i = 0; i < 8; i++) z8[i] = 0;
  __builtin_amdgcn_tensor_load_to_lds(g0, g1, z4, z4, z8, 0);
#else
  (void)lds_off; (void)gp; (void)rows; (void)stride_elems;
#endif
}

// ---------------------------------------------------------------------------
// Tiled GEMM: Y[M=8192][N=1024] = A[M][K=1024] @ W[N][K]^T, (+bias)*scale.
// A and W are bf16. Block tile 128x128, K-step 32, 8 waves, wave tile 32x64.
// Double-buffered TDM staging on TENSORcnt (wave 0 issues; EXEC ignored by TDM).
// EPI 0: bf16 -> (B,NH,S,HD);  EPI 1: bf16 -> (B,NH,HD,S);  EPI 2: f32 row-major
// ---------------------------------------------------------------------------
template <int EPI>
__global__ __launch_bounds__(256) void gemm128(const __bf16* __restrict__ A,
                                               const __bf16* __restrict__ W,
                                               const float* __restrict__ bias,
                                               void* __restrict__ outp,
                                               float scale) {
  __shared__ __bf16 As[2][128 * LDSS];
  __shared__ __bf16 Bs[2][128 * LDSS];
  const int K = H_DIM;
  const int tid = threadIdx.x;
  const int lane = tid & 31, wave = tid >> 5;
  const int lane16 = lane & 15, half = lane >> 4;
  const int m0 = blockIdx.x * 128;
  const int n0 = blockIdx.y * 128;
  const int wm = (wave & 3) * 32;
  const int wn = (wave >> 2) * 64;

  const __bf16* Abase = A + (size_t)m0 * K;
  const __bf16* Wbase = W + (size_t)n0 * K;
  unsigned as_off[2], bs_off[2];
  for (int i = 0; i < 2; i++) {
    as_off[i] = (unsigned)(uintptr_t)(const void*)&As[i][0];
    bs_off[i] = (unsigned)(uintptr_t)(const void*)&Bs[i][0];
  }

  v8f acc[2][4];
  for (int mt = 0; mt < 2; mt++)
    for (int nt = 0; nt < 4; nt++) acc[mt][nt] = zero8();

#if USE_TDM
  if (wave == 0) {                      // prologue: tiles for k0 = 0
    tdm_load_tile(as_off[0], Abase, 128, K);
    tdm_load_tile(bs_off[0], Wbase, 128, K);
  }
  int buf = 0;
  for (int k0 = 0; k0 < K; k0 += 32) {
    if (wave == 0) {
      if (k0 + 32 < K) {                // prefetch next pair into other buffer
        tdm_load_tile(as_off[buf ^ 1], Abase + (k0 + 32), 128, K);
        tdm_load_tile(bs_off[buf ^ 1], Wbase + (k0 + 32), 128, K);
        __builtin_amdgcn_s_wait_tensorcnt(2);   // current pair landed
      } else {
        __builtin_amdgcn_s_wait_tensorcnt(0);
      }
    }
    __syncthreads();
#else
  int buf = 0;
  for (int k0 = 0; k0 < K; k0 += 32) {
    // manual staging fallback: 128x32 bf16 per matrix, 1024 quads each
    for (int i = 0; i < 4; i++) {
      int q = tid + i * 256;
      int row = q >> 3;
      int kq = (q & 7) * 4;
      *(v4bf*)&As[0][row * LDSS + kq] =
          *(const v4bf*)(Abase + (size_t)row * K + k0 + kq);
      *(v4bf*)&Bs[0][row * LDSS + kq] =
          *(const v4bf*)(Wbase + (size_t)row * K + k0 + kq);
    }
    __syncthreads();
#endif

    // A frag (16x32): lane<16 holds K 0..15; lane>=16 K 16..31
    v16bf a[2], b[4];
    for (int mt = 0; mt < 2; mt++) {
      const __bf16* p = &As[buf][(wm + mt * 16 + lane16) * LDSS + half * 16];
      a[mt] = cat16(*(const v8bf*)p, *(const v8bf*)(p + 8));
    }
    // B frag (32x16): lane<16 holds K 0..15 of its column; lane>=16 K 16..31
    for (int nt = 0; nt < 4; nt++) {
      const __bf16* p = &Bs[buf][(wn + nt * 16 + lane16) * LDSS + half * 16];
      b[nt] = cat16(*(const v8bf*)p, *(const v8bf*)(p + 8));
    }
    for (int mt = 0; mt < 2; mt++)
      for (int nt = 0; nt < 4; nt++)
        acc[mt][nt] = __builtin_amdgcn_wmma_f32_16x16x32_bf16(
            false, a[mt], false, b[nt], (short)0, acc[mt][nt], false, false);
    __syncthreads();
    buf ^= 1;
  }

  // Epilogue. C layout: VGPR r holds row (r + 8*half), lane16 is the column.
  for (int mt = 0; mt < 2; mt++) {
    for (int nt = 0; nt < 4; nt++) {
      int n_g = n0 + wn + nt * 16 + lane16;
      float bv = bias[n_g];
      for (int r = 0; r < 8; r++) {
        int m_g = m0 + wm + mt * 16 + r + half * 8;
        float val = (acc[mt][nt][r] + bv) * scale;
        if (EPI == 2) {
          ((float*)outp)[(size_t)m_g * H_DIM + n_g] = val;
        } else {
          int bidx = m_g & (B_SZ - 1);   // m = s*B + b
          int srow = m_g >> 2;
          int hh = n_g >> 6;             // n = h*HD + d
          int dd = n_g & 63;
          size_t idx;
          if (EPI == 0)
            idx = (((size_t)bidx * NH_ + hh) * S_LEN + srow) * HD_ + dd;
          else
            idx = (((size_t)bidx * NH_ + hh) * HD_ + dd) * S_LEN + srow;
          ((__bf16*)outp)[idx] = (__bf16)val;
        }
      }
    }
  }
}

// ---------------------------------------------------------------------------
// Flash attention. Grid (S/128, NH, B), 256 threads = 8 waves, 16 queries/wave.
// Q fragments in registers; K/V fragments straight from global (L2-resident);
// P (f32 C-layout) -> wave-private LDS -> bf16 A-layout for the P@V wmma.
// ---------------------------------------------------------------------------
#define PSTR 72

__global__ __launch_bounds__(256) void flash_attn(
    const __bf16* __restrict__ Q, const __bf16* __restrict__ Kb,
    const __bf16* __restrict__ Vt, const unsigned char* __restrict__ mask,
    __bf16* __restrict__ O) {
  __shared__ __bf16 Pbuf[8 * 16 * PSTR];
  const int lane = threadIdx.x & 31, wave = threadIdx.x >> 5;
  const int lane16 = lane & 15, half = lane >> 4;
  const int b = blockIdx.z, h = blockIdx.y;
  const int q0 = blockIdx.x * 128 + wave * 16;
  const size_t qkbase = (size_t)(b * NH_ + h) * S_LEN * HD_;   // (B,NH,S,HD)
  const size_t vbase  = (size_t)(b * NH_ + h) * HD_ * S_LEN;   // (B,NH,HD,S)
  __bf16* pw = &Pbuf[wave * 16 * PSTR];

  v16bf aq[2];
  for (int ks = 0; ks < 2; ks++) {
    const __bf16* p = Q + qkbase + (size_t)(q0 + lane16) * HD_ + ks * 32 + half * 16;
    aq[ks] = cat16(*(const v8bf*)p, *(const v8bf*)(p + 8));
  }

  v8f o[4];
  for (int nt = 0; nt < 4; nt++) o[nt] = zero8();
  float mrow[8], lrow[8];
  for (int r = 0; r < 8; r++) { mrow[r] = -1.0e30f; lrow[r] = 0.0f; }

  for (int t0 = 0; t0 < S_LEN; t0 += 64) {
    v8f sc[4];
    for (int nt = 0; nt < 4; nt++) {
      sc[nt] = zero8();
      int t = t0 + nt * 16 + lane16;
      for (int ks = 0; ks < 2; ks++) {
        v16bf bk = ldg16(Kb + qkbase + (size_t)t * HD_ + ks * 32 + half * 16);
        sc[nt] = __builtin_amdgcn_wmma_f32_16x16x32_bf16(
            false, aq[ks], false, bk, (short)0, sc[nt], false, false);
      }
    }
    float tmax[8];
    for (int r = 0; r < 8; r++) tmax[r] = -1.0e30f;
    for (int nt = 0; nt < 4; nt++) {
      int t = t0 + nt * 16 + lane16;
      bool msk = mask[(size_t)b * S_LEN + t] != 0;
      for (int r = 0; r < 8; r++) {
        float s = msk ? -1.0e30f : sc[nt][r];
        sc[nt][r] = s;
        tmax[r] = fmaxf(tmax[r], s);
      }
    }
    for (int off = 8; off >= 1; off >>= 1)
      for (int r = 0; r < 8; r++)
        tmax[r] = fmaxf(tmax[r], __shfl_xor(tmax[r], off, 16));

    float alpha[8], psum[8];
    for (int r = 0; r < 8; r++) {
      float mn = fmaxf(mrow[r], tmax[r]);
      alpha[r] = __expf(mrow[r] - mn);
      mrow[r] = mn;
      psum[r] = 0.0f;
    }
    for (int nt = 0; nt < 4; nt++)
      for (int r = 0; r < 8; r++) {
        float e = __expf(sc[nt][r] - mrow[r]);
        sc[nt][r] = e;
        psum[r] += e;
      }
    for (int off = 8; off >= 1; off >>= 1)
      for (int r = 0; r < 8; r++)
        psum[r] += __shfl_xor(psum[r], off, 16);
    for (int r = 0; r < 8; r++) lrow[r] = lrow[r] * alpha[r] + psum[r];

    for (int nt = 0; nt < 4; nt++)
      for (int r = 0; r < 8; r++)
        pw[(r + half * 8) * PSTR + nt * 16 + lane16] = (__bf16)sc[nt][r];

    for (int nt = 0; nt < 4; nt++)
      for (int r = 0; r < 8; r++) o[nt][r] *= alpha[r];

    for (int ks = 0; ks < 2; ks++) {
      const __bf16* pp = pw + lane16 * PSTR + ks * 32 + half * 16;
      v16bf ap = cat16(*(const v8bf*)pp, *(const v8bf*)(pp + 8));
      for (int nt = 0; nt < 4; nt++) {
        int d = nt * 16 + lane16;
        v16bf bv = ldg16(Vt + vbase + (size_t)d * S_LEN + t0 + ks * 32 + half * 16);
        o[nt] = __builtin_amdgcn_wmma_f32_16x16x32_bf16(
            false, ap, false, bv, (short)0, o[nt], false, false);
      }
    }
  }

  for (int nt = 0; nt < 4; nt++) {
    int d = nt * 16 + lane16;
    for (int r = 0; r < 8; r++) {
      int s = q0 + r + half * 8;
      float val = o[nt][r] / lrow[r];
      O[((size_t)s * B_SZ + b) * H_DIM + h * HD_ + d] = (__bf16)val;
    }
  }
}

// ---------------------------------------------------------------------------
extern "C" void kernel_launch(void* const* d_in, const int* in_sizes, int n_in,
                              void* d_out, int out_size, void* d_ws, size_t ws_size,
                              hipStream_t stream) {
  (void)in_sizes; (void)n_in; (void)out_size; (void)ws_size;
  const float* query = (const float*)d_in[0];
  const float* key_t = (const float*)d_in[1];
  const float* value = (const float*)d_in[2];
  const unsigned char* mask = (const unsigned char*)d_in[3];  // numpy bool
  const float* Wq = (const float*)d_in[4];  const float* bq = (const float*)d_in[5];
  const float* Wk = (const float*)d_in[6];  const float* bk = (const float*)d_in[7];
  const float* Wv = (const float*)d_in[8];  const float* bv = (const float*)d_in[9];
  const float* Wo = (const float*)d_in[10]; const float* bo = (const float*)d_in[11];

  const size_t aelem = (size_t)M_TOT * H_DIM;   // 8Mi activation elements
  const size_t welem = (size_t)H_DIM * H_DIM;   // 1Mi weight elements
  __bf16* p = (__bf16*)d_ws;                    // 120 MB total
  __bf16* xq = p; p += aelem;
  __bf16* xk = p; p += aelem;
  __bf16* xv = p; p += aelem;
  __bf16* wqb = p; p += welem;
  __bf16* wkb = p; p += welem;
  __bf16* wvb = p; p += welem;
  __bf16* wob = p; p += welem;
  __bf16* q_ws = p; p += aelem;
  __bf16* k_ws = p; p += aelem;
  __bf16* vt_ws = p; p += aelem;
  __bf16* o_ws = p;

  const int an4 = (int)(aelem / 4), wn4 = (int)(welem / 4);
  cvt_bf16<<<an4 / 256, 256, 0, stream>>>(query, xq, an4);
  cvt_bf16<<<an4 / 256, 256, 0, stream>>>(key_t, xk, an4);
  cvt_bf16<<<an4 / 256, 256, 0, stream>>>(value, xv, an4);
  cvt_bf16<<<wn4 / 256, 256, 0, stream>>>(Wq, wqb, wn4);
  cvt_bf16<<<wn4 / 256, 256, 0, stream>>>(Wk, wkb, wn4);
  cvt_bf16<<<wn4 / 256, 256, 0, stream>>>(Wv, wvb, wn4);
  cvt_bf16<<<wn4 / 256, 256, 0, stream>>>(Wo, wob, wn4);

  dim3 gg(M_TOT / 128, H_DIM / 128);  // (64, 8)
  gemm128<0><<<gg, 256, 0, stream>>>(xq, wqb, bq, (void*)q_ws, 0.125f);
  gemm128<0><<<gg, 256, 0, stream>>>(xk, wkb, bk, (void*)k_ws, 1.0f);
  gemm128<1><<<gg, 256, 0, stream>>>(xv, wvb, bv, (void*)vt_ws, 1.0f);
  flash_attn<<<dim3(S_LEN / 128, NH_, B_SZ), 256, 0, stream>>>(
      q_ws, k_ws, vt_ws, mask, o_ws);
  gemm128<2><<<gg, 256, 0, stream>>>(o_ws, wob, bo, d_out, 1.0f);
}
